// GPTLanguageModel_37580963840267
// MI455X (gfx1250) — matrix operand order repaired and checked by reading, not compiled, not run on hardware
//
#include <hip/hip_runtime.h>

// ---------------- model constants ----------------
#define V_  32000
#define D_  1024
#define S_  1024
#define L_  4
#define DK_ 64
#define DV_ 64
#define H_  16
#define B_  4
#define DFF_ 4096
#define M_  (B_ * S_)          // 4096 tokens

typedef __bf16 bf16_t;
typedef __attribute__((ext_vector_type(16))) __bf16 bf16x16;
typedef __attribute__((ext_vector_type(8)))  float  floatx8;

// ---------------- WMMA helper ----------------
__device__ __forceinline__ floatx8 wmma_bf16(bf16x16 a, bf16x16 b, floatx8 c) {
    // D = A(16x32) * B(32x16) + C ; f32 accumulate
    return __builtin_amdgcn_wmma_f32_16x16x32_bf16(
        /*neg_a=*/false, a, /*neg_b=*/false, b,
        /*c_mod=*/(short)0, c, /*reuse_a=*/false, /*reuse_b=*/false);
}

union AFragU { bf16x16 v; uint4 u[2]; };

// A fragment (16x32, bf16, row-major source with leading dim lda):
// lane<16 : row=m0+lane,   K = k0+0..15
// lane>=16: row=m0+lane-16,K = k0+16..31
__device__ __forceinline__ bf16x16 load_a_frag(const bf16_t* A, int lda, int m0,
                                               int k0, int lane) {
    int row = m0 + (lane & 15);
    int kb  = k0 + 16 * (lane >> 4);
    const bf16_t* p = A + (long)row * lda + kb;
    AFragU f;
    f.u[0] = *(const uint4*)(p);
    f.u[1] = *(const uint4*)(p + 8);
    return f.v;
}

// B fragment from packed-tile layout produced by pack_w_kernel:
// element e of lane l in tile (kt,nt) = B[kt*32 + 16*(l>>4) + e][nt*16 + (l&15)]
__device__ __forceinline__ bf16x16 load_b_packed(const bf16_t* P, int KT, int nt,
                                                 int kt, int lane) {
    const bf16_t* p = P + (((long)nt * KT + kt) * 32 + lane) * 16;
    AFragU f;
    f.u[0] = *(const uint4*)(p);
    f.u[1] = *(const uint4*)(p + 8);
    return f.v;
}

// ---------------- embedding ----------------
__global__ void embed_kernel(const int* __restrict__ idx,
                             const float* __restrict__ tok,
                             const float* __restrict__ pos,
                             float* __restrict__ x) {
    int i = blockIdx.x * 256 + threadIdx.x;              // over M_*D_
    if (i >= M_ * D_) return;
    int m = i >> 10;                                     // D_ == 1024
    int d = i & 1023;
    int t = idx[m];
    x[i] = tok[(long)t * D_ + d] + pos[(long)(m & (S_ - 1)) * D_ + d];
}

// ---------------- layernorm (f32 in -> bf16 out) ----------------
__global__ void ln_kernel(const float* __restrict__ x,
                          const float* __restrict__ g,
                          const float* __restrict__ b,
                          bf16_t* __restrict__ out) {
    int row  = blockIdx.x;
    int lane = threadIdx.x & 31;
    int wave = threadIdx.x >> 5;
    const float* xr = x + (long)row * D_;
    float vals[4];
    float s = 0.f, ss = 0.f;
#pragma unroll
    for (int i = 0; i < 4; ++i) {
        float v = xr[threadIdx.x + 256 * i];
        vals[i] = v; s += v; ss += v * v;
    }
#pragma unroll
    for (int off = 16; off > 0; off >>= 1) {
        s  += __shfl_down(s, off, 32);
        ss += __shfl_down(ss, off, 32);
    }
    __shared__ float sh_s[8], sh_q[8];
    if (lane == 0) { sh_s[wave] = s; sh_q[wave] = ss; }
    __syncthreads();
    if (threadIdx.x == 0) {
        float a = 0.f, q = 0.f;
#pragma unroll
        for (int i = 0; i < 8; ++i) { a += sh_s[i]; q += sh_q[i]; }
        sh_s[0] = a; sh_q[0] = q;
    }
    __syncthreads();
    float mean = sh_s[0] * (1.f / D_);
    float var  = sh_q[0] * (1.f / D_) - mean * mean;
    float rstd = rsqrtf(var + 1e-5f);
#pragma unroll
    for (int i = 0; i < 4; ++i) {
        int c = threadIdx.x + 256 * i;
        out[(long)row * D_ + c] = (bf16_t)((vals[i] - mean) * rstd * g[c] + b[c]);
    }
}

// ---------------- weight pack: f32 -> bf16 WMMA tile layout ----------------
// dst[((nt*KT + kt)*32 + lane)*16 + e] = B[k][n],
//   k = kt*32 + 16*(lane>>4) + e, n = nt*16 + (lane&15)
// qkv==1: source is [H][K][64] and logical B[k][n] = src[n>>6][k][n&63]
__global__ void pack_w_kernel(const float* __restrict__ src,
                              bf16_t* __restrict__ dst,
                              int K, int N, int qkv) {
    long i = (long)blockIdx.x * 256 + threadIdx.x;
    long total = (long)K * N;
    if (i >= total) return;
    int  e    = (int)(i & 15);
    long t    = i >> 4;
    int  lane = (int)(t & 31);
    long tile = t >> 5;
    int  KT   = K >> 5;
    int  nt   = (int)(tile / KT);
    int  kt   = (int)(tile % KT);
    int  k    = kt * 32 + 16 * (lane >> 4) + e;
    int  n    = nt * 16 + (lane & 15);
    float v = qkv ? src[(long)(n >> 6) * K * 64 + (long)k * 64 + (n & 63)]
                  : src[(long)k * N + n];
    dst[i] = (bf16_t)v;
}

// ---------------- generic WMMA GEMM with templated fused epilogue ----------
// C[M_ x N] = A[M_ x K](bf16) * Bp(packed bf16)
// HAS_BIAS: add bias[N]; RELU: relu; HAS_RES: add resid f32; OUT_BF16: dtype
template <bool HAS_BIAS, bool RELU, bool HAS_RES, bool OUT_BF16>
__global__ void gemm_kernel(const bf16_t* __restrict__ A,
                            const bf16_t* __restrict__ Bp,
                            void* __restrict__ Cout,
                            const float* __restrict__ bias,
                            const float* __restrict__ resid,
                            int N, int K) {
    int lane = threadIdx.x & 31;
    int wave = threadIdx.x >> 5;
    int wm = wave & 1, wn = wave >> 1;
    int m_base = blockIdx.y * 64  + wm * 32;
    int n_base = blockIdx.x * 256 + wn * 64;
    int KT = K >> 5;

    floatx8 acc[2][4];
#pragma unroll
    for (int i = 0; i < 2; ++i)
#pragma unroll
        for (int j = 0; j < 4; ++j)
            acc[i][j] = (floatx8){0.f, 0.f, 0.f, 0.f, 0.f, 0.f, 0.f, 0.f};

    for (int k0 = 0; k0 < K; k0 += 32) {
        bf16x16 a0 = load_a_frag(A, K, m_base,      k0, lane);
        bf16x16 a1 = load_a_frag(A, K, m_base + 16, k0, lane);
        int kt = k0 >> 5;
#pragma unroll
        for (int j = 0; j < 4; ++j) {
            bf16x16 bfr = load_b_packed(Bp, KT, (n_base >> 4) + j, kt, lane);
            acc[0][j] = wmma_bf16(a0, bfr, acc[0][j]);
            acc[1][j] = wmma_bf16(a1, bfr, acc[1][j]);
        }
    }

    int hi = lane >> 4, lo = lane & 15;
    float*  Cf = (float*)Cout;
    bf16_t* Cb = (bf16_t*)Cout;
#pragma unroll
    for (int i = 0; i < 2; ++i)
#pragma unroll
        for (int j = 0; j < 4; ++j) {
            int n = n_base + j * 16 + lo;
            float bv = HAS_BIAS ? bias[n] : 0.f;
#pragma unroll
            for (int r = 0; r < 8; ++r) {
                int m = m_base + i * 16 + r + 8 * hi;
                float v = acc[i][j][r] + bv;
                if (RELU) v = fmaxf(v, 0.f);
                long o = (long)m * N + n;
                if (HAS_RES) v += resid[o];
                if (OUT_BF16) Cb[o] = (bf16_t)v;
                else          Cf[o] = v;
            }
        }
}

// ---------------- V transpose: [b,t,h,dv] -> [b,h,dv,t] ----------------
__global__ void vtrans_kernel(const bf16_t* __restrict__ v,
                              bf16_t* __restrict__ vt) {
    long i = (long)blockIdx.x * 256 + threadIdx.x;   // over B_*H_*DV_*S_
    if (i >= (long)B_ * H_ * DV_ * S_) return;
    int t    = (int)(i & (S_ - 1));
    long rst = i >> 10;
    int dv   = (int)(rst & 63); rst >>= 6;
    int h    = (int)(rst & 15);
    int b    = (int)(rst >> 4);
    vt[i] = v[(((long)b * S_ + t) << 10) + h * 64 + dv];
}

// ---------------- causal flash attention (one wave = one 16-row Q tile) ----
__global__ void attn_kernel(const bf16_t* __restrict__ Q,
                            const bf16_t* __restrict__ Kx,
                            const bf16_t* __restrict__ Vt,
                            bf16_t* __restrict__ O) {
    int lane = threadIdx.x & 31;
    int wave = threadIdx.x >> 5;
    int bh = blockIdx.x;
    int b  = bh >> 4, h = bh & 15;
    int qt = blockIdx.y * 8 + wave;
    int q0 = qt * 16;
    int hi = lane >> 4, lo = lane & 15;

    __shared__ bf16_t pbuf[8][16 * 32];
    bf16_t* myP = pbuf[wave];

    const bf16_t* Qb = Q  + ((long)b * S_) * (H_ * DK_) + h * DK_;
    const bf16_t* Kb = Kx + ((long)b * S_) * (H_ * DK_) + h * DK_;
    const bf16_t* Vb = Vt + ((long)(b * H_ + h) * DV_) * S_;   // [dv][t]

    // Q fragments are invariant across the key loop: hoist (2 x 8 VGPRs)
    AFragU aq[2];
#pragma unroll
    for (int dsl = 0; dsl < 2; ++dsl) {
        const bf16_t* ap = Qb + (long)(q0 + lo) * (H_ * DK_) + dsl * 32 + 16 * hi;
        aq[dsl].u[0] = *(const uint4*)ap;
        aq[dsl].u[1] = *(const uint4*)(ap + 8);
    }

    floatx8 oacc[4];
#pragma unroll
    for (int n = 0; n < 4; ++n)
        oacc[n] = (floatx8){0.f, 0.f, 0.f, 0.f, 0.f, 0.f, 0.f, 0.f};
    float mrow[8], lrow[8];
#pragma unroll
    for (int r = 0; r < 8; ++r) { mrow[r] = -3.0e38f; lrow[r] = 0.f; }

    const float scale = 0.125f;   // 1/sqrt(64)

    for (int kb = 0; kb <= q0 + 15; kb += 32) {
        floatx8 s[2];
        s[0] = (floatx8){0.f, 0.f, 0.f, 0.f, 0.f, 0.f, 0.f, 0.f};
        s[1] = s[0];
#pragma unroll
        for (int half = 0; half < 2; ++half) {
            int t0 = kb + half * 16;
#pragma unroll
            for (int dsl = 0; dsl < 2; ++dsl) {
                // B frag: K^T, contiguous in d per lane
                const bf16_t* bp = Kb + (long)(t0 + lo) * (H_ * DK_) + dsl * 32 + 16 * hi;
                AFragU bfr; bfr.u[0] = *(const uint4*)bp; bfr.u[1] = *(const uint4*)(bp + 8);
                s[half] = wmma_bf16(aq[dsl].v, bfr.v, s[half]);
            }
        }
        // mask + online softmax
        float pv[2][8];
#pragma unroll
        for (int r = 0; r < 8; ++r) {
            int qrow = q0 + r + 8 * hi;
            float s0 = s[0][r] * scale;
            float s1 = s[1][r] * scale;
            if (kb + lo      > qrow) s0 = -3.0e38f;
            if (kb + 16 + lo > qrow) s1 = -3.0e38f;
            float mx = fmaxf(s0, s1);
#pragma unroll
            for (int off = 8; off >= 1; off >>= 1)
                mx = fmaxf(mx, __shfl_xor(mx, off, 32));
            float mnew = fmaxf(mrow[r], mx);
            float p0 = __expf(s0 - mnew);
            float p1 = __expf(s1 - mnew);
            float ps = p0 + p1;
#pragma unroll
            for (int off = 8; off >= 1; off >>= 1)
                ps += __shfl_xor(ps, off, 32);
            float alpha = __expf(mrow[r] - mnew);
            lrow[r] = lrow[r] * alpha + ps;
            mrow[r] = mnew;
            pv[0][r] = p0; pv[1][r] = p1;
#pragma unroll
            for (int n = 0; n < 4; ++n) oacc[n][r] *= alpha;
        }
        // stage P (16x32) into LDS in row-major, reload as A fragment
#pragma unroll
        for (int half = 0; half < 2; ++half)
#pragma unroll
            for (int r = 0; r < 8; ++r)
                myP[(r + 8 * hi) * 32 + half * 16 + lo] = (bf16_t)pv[half][r];
        const bf16_t* pp = myP + lo * 32 + 16 * hi;
        AFragU pf; pf.u[0] = *(const uint4*)pp; pf.u[1] = *(const uint4*)(pp + 8);
        // P(16x32) x V(32x64)
#pragma unroll
        for (int n = 0; n < 4; ++n) {
            const bf16_t* vp = Vb + (long)(n * 16 + lo) * S_ + kb + 16 * hi;
            AFragU vf; vf.u[0] = *(const uint4*)vp; vf.u[1] = *(const uint4*)(vp + 8);
            oacc[n] = wmma_bf16(pf.v, vf.v, oacc[n]);
        }
    }
    // write O[b, q, h, dv] bf16
    bf16_t* Ob = O + ((long)b * S_) * (H_ * DV_) + h * DV_;
#pragma unroll
    for (int n = 0; n < 4; ++n)
#pragma unroll
        for (int r = 0; r < 8; ++r) {
            float val = oacc[n][r] / lrow[r];
            Ob[(long)(q0 + r + 8 * hi) * (H_ * DV_) + n * 16 + lo] = (bf16_t)val;
        }
}

// ---------------- host orchestration ----------------
extern "C" void kernel_launch(void* const* d_in, const int* in_sizes, int n_in,
                              void* d_out, int out_size, void* d_ws, size_t ws_size,
                              hipStream_t stream) {
    (void)in_sizes; (void)n_in; (void)out_size; (void)ws_size;
    const int*   idx     = (const int*)  d_in[0];
    const float* tok_emb = (const float*)d_in[1];
    const float* pos_emb = (const float*)d_in[2];
    const float* wq      = (const float*)d_in[3];
    const float* wk      = (const float*)d_in[4];
    const float* wv      = (const float*)d_in[5];
    const float* wo      = (const float*)d_in[6];
    const float* w1      = (const float*)d_in[7];
    const float* b1      = (const float*)d_in[8];
    const float* w2      = (const float*)d_in[9];
    const float* b2      = (const float*)d_in[10];
    const float* ln1_g   = (const float*)d_in[11];
    const float* ln1_b   = (const float*)d_in[12];
    const float* ln2_g   = (const float*)d_in[13];
    const float* ln2_b   = (const float*)d_in[14];
    const float* lnf_g   = (const float*)d_in[15];
    const float* lnf_b   = (const float*)d_in[16];
    const float* lm_w    = (const float*)d_in[17];
    const float* lm_b    = (const float*)d_in[18];
    float* logits = (float*)d_out;

    // workspace carve-up (256B aligned)
    char*  ws  = (char*)d_ws;
    size_t off = 0;
    auto alloc = [&](size_t bytes) -> void* {
        off = (off + 255) & ~(size_t)255;
        void* p = ws + off;
        off += bytes;
        return p;
    };
    float*  x   = (float*) alloc((size_t)M_ * D_ * 4);
    bf16_t* hb  = (bf16_t*)alloc((size_t)M_ * D_ * 2);          // LN output
    bf16_t* qb  = (bf16_t*)alloc((size_t)M_ * H_ * DK_ * 2);
    bf16_t* kbf = (bf16_t*)alloc((size_t)M_ * H_ * DK_ * 2);
    bf16_t* vb  = (bf16_t*)alloc((size_t)M_ * H_ * DV_ * 2);
    bf16_t* vtb = (bf16_t*)alloc((size_t)B_ * H_ * DV_ * S_ * 2);
    bf16_t* ob  = (bf16_t*)alloc((size_t)M_ * H_ * DV_ * 2);
    bf16_t* tb  = (bf16_t*)alloc((size_t)M_ * DFF_ * 2);
    bf16_t* pw  = (bf16_t*)alloc((size_t)D_ * V_ * 2);          // packed weight (reused)

    dim3 blk(256);

    auto pack = [&](const float* Wsrc, int K, int N, int qkv) {
        long tot = (long)K * N;
        pack_w_kernel<<<dim3((unsigned)((tot + 255) / 256)), blk, 0, stream>>>(
            Wsrc, pw, K, N, qkv);
    };
    auto ggrid = [](int N) { return dim3(N / 256, M_ / 64); };

    // x = tok_emb[idx] + pos_emb
    embed_kernel<<<dim3((M_ * D_ + 255) / 256), blk, 0, stream>>>(idx, tok_emb, pos_emb, x);

    for (int l = 0; l < L_; ++l) {
        const float* wq_l = wq + (long)l * H_ * D_ * DK_;
        const float* wk_l = wk + (long)l * H_ * D_ * DK_;
        const float* wv_l = wv + (long)l * H_ * D_ * DV_;
        const float* wo_l = wo + (long)l * (H_ * DV_) * D_;
        const float* w1_l = w1 + (long)l * D_ * DFF_;
        const float* w2_l = w2 + (long)l * DFF_ * D_;

        // h = LN1(x)
        ln_kernel<<<dim3(M_), blk, 0, stream>>>(x, ln1_g + (long)l * D_, ln1_b + (long)l * D_, hb);

        // Q,K,V projections (bf16 out, no epilogue)
        pack(wq_l, D_, H_ * DK_, 1);
        gemm_kernel<false, false, false, true><<<ggrid(H_ * DK_), blk, 0, stream>>>(
            hb, pw, qb, nullptr, nullptr, H_ * DK_, D_);
        pack(wk_l, D_, H_ * DK_, 1);
        gemm_kernel<false, false, false, true><<<ggrid(H_ * DK_), blk, 0, stream>>>(
            hb, pw, kbf, nullptr, nullptr, H_ * DK_, D_);
        pack(wv_l, D_, H_ * DV_, 1);
        gemm_kernel<false, false, false, true><<<ggrid(H_ * DV_), blk, 0, stream>>>(
            hb, pw, vb, nullptr, nullptr, H_ * DV_, D_);

        // V -> [b,h,dv,t]
        vtrans_kernel<<<dim3((B_ * H_ * DV_ * S_ + 255) / 256), blk, 0, stream>>>(vb, vtb);
        // causal attention
        attn_kernel<<<dim3(B_ * H_, S_ / 128), blk, 0, stream>>>(qb, kbf, vtb, ob);

        // x = x + O @ wo   (residual fused, f32 out, in-place)
        pack(wo_l, H_ * DV_, D_, 0);
        gemm_kernel<false, false, true, false><<<ggrid(D_), blk, 0, stream>>>(
            ob, pw, x, nullptr, x, D_, H_ * DV_);

        // h = LN2(x)
        ln_kernel<<<dim3(M_), blk, 0, stream>>>(x, ln2_g + (long)l * D_, ln2_b + (long)l * D_, hb);

        // t = relu(h @ w1 + b1)  (bf16 out)
        pack(w1_l, D_, DFF_, 0);
        gemm_kernel<true, true, false, true><<<ggrid(DFF_), blk, 0, stream>>>(
            hb, pw, tb, b1 + (long)l * DFF_, nullptr, DFF_, D_);

        // x = x + t @ w2 + b2   (f32 out, in-place)
        pack(w2_l, DFF_, D_, 0);
        gemm_kernel<true, false, true, false><<<ggrid(D_), blk, 0, stream>>>(
            tb, pw, x, b2 + (long)l * D_, x, D_, DFF_);
    }

    // final LN, LM head
    ln_kernel<<<dim3(M_), blk, 0, stream>>>(x, lnf_g, lnf_b, hb);
    pack(lm_w, D_, V_, 0);
    gemm_kernel<true, false, false, false><<<ggrid(V_), blk, 0, stream>>>(
        hb, pw, logits, lm_b, nullptr, V_, D_);
}
